// MoEAdapter_55774445306355
// MI455X (gfx1250) — matrix operand order repaired and checked by reading, not compiled, not run on hardware
//
#include <hip/hip_runtime.h>
#include <stdint.h>

// ---------------------------------------------------------------------------
// MoE low-rank adapter, fused, bf16 WMMA (gfx1250 / CDNA5, wave32)
// B=8 S=4096 D=1024 E=8 R=16 K=2.  N = 32768 tokens.
// ---------------------------------------------------------------------------

typedef __attribute__((ext_vector_type(16))) __bf16 v16bf;
typedef __attribute__((ext_vector_type(8)))  float  v8f;

#define D_DIM   1024
#define E_EXP   8
#define R_RANK  16
#define ER      128          // E*R, down-GEMM N / up-GEMM K
#define MT      64           // tokens per block
#define XPITCH  (D_DIM + 8)  // bf16 units; +8 keeps 16B align, breaks bank conflicts
#define HPITCH  (ER + 8)
#define WAUGK   32           // bias k-step (8 real + 24 zero)
#define WAUGP   40

__device__ __forceinline__ uint16_t f2bf(float f) {
  uint32_t u = __builtin_bit_cast(uint32_t, f);
  uint32_t r = u + 0x7FFFu + ((u >> 16) & 1u);   // round-to-nearest-even
  return (uint16_t)(r >> 16);
}

// A fragment (16-bit A 16x32 layout): lane's row, chunks at k=8*half, 16+8*half
__device__ __forceinline__ v16bf load_a(const uint16_t* row, int kbase, int half) {
  union { v16bf v; uint4 q[2]; } a;
  a.q[0] = *(const uint4*)(row + kbase + 8 * half);
  a.q[1] = *(const uint4*)(row + kbase + 16 + 8 * half);
  return a.v;
}

// B fragment (16-bit B 32x16): lane's column (pre-transposed, k-contiguous),
// chunks at k = kbase + 8*half, kbase + 16 + 8*half
__device__ __forceinline__ v16bf load_b(const uint16_t* col, int kbase, int half) {
  union { v16bf v; uint4 q[2]; } b;
  b.q[0] = *(const uint4*)(col + kbase + 8 * half);
  b.q[1] = *(const uint4*)(col + kbase + 16 + 8 * half);
  return b.v;
}

__device__ __forceinline__ v8f wmma_bf16(v16bf a, v16bf b, v8f c) {
  return __builtin_amdgcn_wmma_f32_16x16x32_bf16(false, a, false, b, (short)0, c,
                                                 false, false);
}

// ---------------------------------------------------------------------------
// Prep: convert + transpose weights to bf16 in workspace.
//   Wrt [16][1024]  (cols 8..15 zero)        Wrt[n*1024+k] = Wr[k][n]
//   Wdt [128][1024]                          Wdt[c*1024+k] = Wd[e][k][r], c=e*16+r
//   Wut [1024][128]                          Wut[d*128+k]  = Wu[k/16][k%16][d]
//   but [1024][32]   (k 8..31 zero)          but[d*32+k]   = bu[k][d]
// ---------------------------------------------------------------------------
__global__ void moe_prep_kernel(const float* __restrict__ Wr,
                                const float* __restrict__ Wd,
                                const float* __restrict__ Wu,
                                const float* __restrict__ bu,
                                uint16_t* __restrict__ Wrt,
                                uint16_t* __restrict__ Wdt,
                                uint16_t* __restrict__ Wut,
                                uint16_t* __restrict__ but) {
  int idx = blockIdx.x * 256 + threadIdx.x;
  const int nWrt = 16 * D_DIM, nWdt = ER * D_DIM, nWut = D_DIM * ER, nbut = D_DIM * WAUGK;
  if (idx < nWrt) {
    int n = idx >> 10, k = idx & 1023;
    Wrt[idx] = (n < E_EXP) ? f2bf(Wr[k * E_EXP + n]) : (uint16_t)0;
    return;
  }
  idx -= nWrt;
  if (idx < nWdt) {
    int c = idx >> 10, k = idx & 1023;
    int e = c >> 4, r = c & 15;
    Wdt[idx] = f2bf(Wd[(e * D_DIM + k) * R_RANK + r]);
    return;
  }
  idx -= nWdt;
  if (idx < nWut) {
    int d = idx >> 7, k = idx & 127;
    Wut[idx] = f2bf(Wu[k * D_DIM + d]);   // (e*16+r)*D + d == k*D + d
    return;
  }
  idx -= nWut;
  if (idx < nbut) {
    int d = idx >> 5, k = idx & 31;
    but[idx] = (k < E_EXP) ? f2bf(bu[k * D_DIM + d]) : (uint16_t)0;
  }
}

// ---------------------------------------------------------------------------
// Main fused kernel: 256 threads (8 waves), 64 tokens per block.
//   wave -> (mtile = wave>>1, chalf = wave&1); mtile picks 16-token M tile.
// ---------------------------------------------------------------------------
__global__ __launch_bounds__(256)
void moe_main_kernel(const float* __restrict__ x,
                     const float* __restrict__ br,
                     const float* __restrict__ bd,
                     const uint16_t* __restrict__ Wrt,
                     const uint16_t* __restrict__ Wdt,
                     const uint16_t* __restrict__ Wut,
                     const uint16_t* __restrict__ but,
                     float* __restrict__ out) {
  __shared__ uint16_t xb[MT * XPITCH];   // x tile, bf16            ~129 KB
  __shared__ uint16_t hb[MT * HPITCH];   // gated relu(down) bf16   ~17 KB
  __shared__ uint16_t waug[MT * WAUGP];  // [w|0] bias A tile        ~5 KB
  __shared__ float    lgt[MT * 16];      // router logits            4 KB
  __shared__ float    wg[MT * E_EXP];    // gate weights             2 KB
  __shared__ float    bdl[ER];           // down bias               .5 KB

  const int tid  = threadIdx.x;
  const int lane = tid & 31;
  const int half = lane >> 4;
  const int ln   = lane & 15;
  const int wave = tid >> 5;
  const int mtile = wave >> 1;
  const int chalf = wave & 1;
  const long tok0 = (long)blockIdx.x * MT;

  if (tid < ER) bdl[tid] = bd[tid];

  // ---- Phase 0: stage x tile -> bf16 LDS (x read exactly once from HBM) ----
  {
    int row = tid >> 2, seg = tid & 3;                 // 4 threads per row
    const float4* src = (const float4*)(x + (size_t)(tok0 + row) * D_DIM) + seg * 64;
    uint16_t* dst = xb + row * XPITCH + seg * 256;
    for (int i = 0; i < 64; ++i) {
      float4 f = src[i];
      uint2 p;
      p.x = (uint32_t)f2bf(f.x) | ((uint32_t)f2bf(f.y) << 16);
      p.y = (uint32_t)f2bf(f.z) | ((uint32_t)f2bf(f.w) << 16);
      *(uint2*)(dst + i * 4) = p;
    }
  }
  __syncthreads();

  const uint16_t* arow = xb + (mtile * 16 + ln) * XPITCH;  // per-lane A row

  // ---- Phase 1: router GEMM [16x1024]x[1024x16] (E padded to 16) ----
  if (chalf == 0) {
    const uint16_t* bcol = Wrt + ln * D_DIM;
    v8f acc = {};
    for (int k0 = 0; k0 < D_DIM; k0 += 32)
      acc = wmma_bf16(load_a(arow, k0, half), load_b(bcol, k0, half), acc);
    for (int j = 0; j < 8; ++j)
      lgt[(mtile * 16 + j + 8 * half) * 16 + ln] = acc[j];
  }
  __syncthreads();

  // ---- Phase 2: softmax + top-2 + renorm (one thread per token) ----
  if (tid < MT) {
    float l[E_EXP], mx = -1e30f;
    for (int e = 0; e < E_EXP; ++e) { l[e] = lgt[tid * 16 + e] + br[e]; mx = fmaxf(mx, l[e]); }
    for (int e = 0; e < E_EXP; ++e) l[e] = __expf(l[e] - mx);   // softmax denom cancels
    int i0 = 0; float v0 = l[0];
    for (int e = 1; e < E_EXP; ++e) if (l[e] > v0) { v0 = l[e]; i0 = e; }
    int i1 = -1; float v1 = -1.0f;
    for (int e = 0; e < E_EXP; ++e) if (e != i0 && l[e] > v1) { v1 = l[e]; i1 = e; }
    float inv = 1.0f / (v0 + v1);
    for (int e = 0; e < E_EXP; ++e) {
      float w = (e == i0) ? v0 * inv : ((e == i1) ? v1 * inv : 0.0f);
      wg[tid * E_EXP + e]   = w;
      waug[tid * WAUGP + e] = f2bf(w);
    }
    for (int k = E_EXP; k < WAUGK; ++k) waug[tid * WAUGP + k] = 0;
  }
  __syncthreads();

  // ---- Phase 3: down GEMM [16x1024]x[1024x128] + bias + ReLU + gate ----
  for (int nt = 0; nt < 4; ++nt) {
    const int col = (chalf * 4 + nt) * 16 + ln;         // 0..127 -> (e,r)
    const uint16_t* bcol = Wdt + col * D_DIM;
    v8f acc = {};
    for (int k0 = 0; k0 < D_DIM; k0 += 32) {
      if (k0 + 128 < D_DIM) __builtin_prefetch(bcol + k0 + 128, 0, 1);
      acc = wmma_bf16(load_a(arow, k0, half), load_b(bcol, k0, half), acc);
    }
    const int   e   = col >> 4;
    const float bdv = bdl[col];
    for (int j = 0; j < 8; ++j) {
      int m = mtile * 16 + j + 8 * half;
      float h = acc[j] + bdv;
      h = h > 0.0f ? h : 0.0f;
      h *= wg[m * E_EXP + e];
      hb[m * HPITCH + col] = f2bf(h);
    }
  }
  __syncthreads();

  // ---- Phase 4: up GEMM [16x128]x[128x1024] + one bias WMMA k-step ----
  const uint16_t* hrow = hb   + (mtile * 16 + ln) * HPITCH;
  const uint16_t* wrow = waug + (mtile * 16 + ln) * WAUGP;
  for (int nt = 0; nt < 32; ++nt) {
    const int dcol = (chalf * 32 + nt) * 16 + ln;       // output column d
    const uint16_t* bcol = Wut + dcol * ER;
    v8f acc = {};
    for (int k0 = 0; k0 < ER; k0 += 32)
      acc = wmma_bf16(load_a(hrow, k0, half), load_b(bcol, k0, half), acc);
    // gated bias: [w | 0]_{16x32} x [bu^T | 0]_{32x16}
    acc = wmma_bf16(load_a(wrow, 0, half), load_b(but + dcol * WAUGK, 0, half), acc);
    for (int j = 0; j < 8; ++j) {
      long m = tok0 + mtile * 16 + j + 8 * half;
      out[(size_t)m * D_DIM + dcol] = acc[j];
    }
  }
}

// ---------------------------------------------------------------------------
extern "C" void kernel_launch(void* const* d_in, const int* in_sizes, int n_in,
                              void* d_out, int out_size, void* d_ws, size_t ws_size,
                              hipStream_t stream) {
  const float* x  = (const float*)d_in[0];
  const float* Wr = (const float*)d_in[1];
  const float* br = (const float*)d_in[2];
  const float* Wd = (const float*)d_in[3];
  const float* bd = (const float*)d_in[4];
  const float* Wu = (const float*)d_in[5];
  const float* bu = (const float*)d_in[6];

  uint16_t* ws  = (uint16_t*)d_ws;
  uint16_t* Wrt = ws;                          // 16*1024
  uint16_t* Wdt = Wrt + 16 * D_DIM;            // 128*1024
  uint16_t* Wut = Wdt + ER * D_DIM;            // 1024*128
  uint16_t* but = Wut + D_DIM * ER;            // 1024*32

  const int total = 16 * D_DIM + ER * D_DIM + D_DIM * ER + D_DIM * WAUGK;
  moe_prep_kernel<<<(total + 255) / 256, 256, 0, stream>>>(Wr, Wd, Wu, bu,
                                                           Wrt, Wdt, Wut, but);

  const int ntok = 8 * 4096;                   // B*S
  moe_main_kernel<<<ntok / MT, 256, 0, stream>>>(x, br, bd, Wrt, Wdt, Wut, but,
                                                 (float*)d_out);
}
